// HemeNet_73701638800039
// MI455X (gfx1250) — matrix operand order repaired and checked by reading, not compiled, not run on hardware
//
#include <hip/hip_runtime.h>

#define Nn 10000
#define Ee 160000
#define Cc 4
#define CFf 16
#define Rr 7
#define Hh 256
#define Ll 2
#define KE1 576   /* 560 padded to 32-multiple */
#define KN1 2048

typedef __bf16 bf16_t;
typedef __attribute__((ext_vector_type(16))) __bf16 v16bf;
typedef __attribute__((ext_vector_type(8)))  __bf16 bf16x8;
typedef __attribute__((ext_vector_type(8)))  float  v8f;

__device__ __forceinline__ float silu_f(float x) { return x / (1.0f + __expf(-x)); }

__device__ __forceinline__ v8f wmma_bf16(v16bf a, v16bf b, v8f c) {
#if defined(__AMDGCN__)
  return __builtin_amdgcn_wmma_f32_16x16x32_bf16(false, a, false, b, (short)0, c, false, false);
#else
  return c;
#endif
}

// Combine two aligned 8-half LDS groups into one fragment register set.
__device__ __forceinline__ v16bf ld_frag_pair(const bf16_t* p0, const bf16_t* p1) {
  bf16x8 lo = *(const bf16x8*)p0;
  bf16x8 hi = *(const bf16x8*)p1;
  return __builtin_shufflevector(lo, hi, 0, 1, 2, 3, 4, 5, 6, 7, 8, 9, 10, 11, 12, 13, 14, 15);
}

// A fragment: 16x32 (MxK) tile, LDS row-major, row stride 40 halves (80B, 16B-aligned).
// ISA layout: lanes 0-15 M=lane, K 0..7 & 16..23; lanes 16-31 K 8..15 & 24..31.
__device__ __forceinline__ v16bf load_a_frag(const bf16_t* As, int lane) {
  int m = lane & 15, kq = lane >> 4;
  const bf16_t* p0 = As + m * 40 + kq * 8;
  return ld_frag_pair(p0, p0 + 16);
}

// B fragment: 32x16 (KxN) tile from K-contiguous (transposed) LDS WsT[n][k],
// row stride 40 halves. Lane holds N=n0+(lane&15), K = kq*8 + j (j<8), 16 + kq*8 + (j-8) (j>=8).
__device__ __forceinline__ v16bf load_b_frag(const bf16_t* WsT, int n0, int lane) {
  int n = n0 + (lane & 15), kq = lane >> 4;
  const bf16_t* p0 = WsT + n * 40 + kq * 8;
  return ld_frag_pair(p0, p0 + 16);
}

// ---------------- utility kernels ----------------

__global__ void k_zero(float* __restrict__ p, long long n) {
  long long i = (long long)blockIdx.x * blockDim.x + threadIdx.x;
  long long stride = (long long)gridDim.x * blockDim.x;
  for (; i < n; i += stride) p[i] = 0.0f;
}

// dst (Ncols x K, i.e. W^T) from src (K x Ncols row-major), one (matrix,layer) per launch
__global__ void k_cvtT(const float* __restrict__ src, bf16_t* __restrict__ dst, int K, int Ncols) {
  long long i = (long long)blockIdx.x * blockDim.x + threadIdx.x;
  if (i >= (long long)K * Ncols) return;
  int n = (int)(i / K), k = (int)(i - (long long)n * K);
  dst[i] = (bf16_t)src[(size_t)k * Ncols + n];
}

// W_e1 (560x256) -> transposed, K padded: dst (256 x 576) bf16
__global__ void k_pad_we1T(const float* __restrict__ src, bf16_t* __restrict__ dst) {
  int i = blockIdx.x * blockDim.x + threadIdx.x;  // over 256*576
  if (i >= 256 * KE1) return;
  int n = i / KE1, k = i - n * KE1;
  dst[i] = (k < 560) ? (bf16_t)src[k * 256 + n] : (bf16_t)0.0f;
}

// ca[n,f] = sum_c attr[n,c,f]*cw[n,c] / sum_c cw[n,c]
__global__ void k_ca(const float* __restrict__ attr, const float* __restrict__ cw,
                     float* __restrict__ ca) {
  int n = blockIdx.x * blockDim.x + threadIdx.x;
  if (n >= Nn) return;
  float cs = 0.0f;
  float w[Cc];
#pragma unroll
  for (int c = 0; c < Cc; ++c) { w[c] = cw[n * Cc + c]; cs += w[c]; }
  float inv = 1.0f / cs;
#pragma unroll
  for (int f = 0; f < CFf; ++f) {
    float s = 0.0f;
#pragma unroll
    for (int c = 0; c < Cc; ++c) s += attr[(n * Cc + c) * CFf + f] * w[c];
    ca[n * CFf + f] = s * inv;
  }
}

// Per edge: xd (E x 4 x 4 x 3), radial*wmask, rad_feat = silu(radial @ W_rad + b_rad)
__global__ void k_edge_geom(const int* __restrict__ el, const float* __restrict__ x,
                            const float* __restrict__ cw,
                            const float* __restrict__ Wr, const float* __restrict__ br,
                            float* __restrict__ xd, bf16_t* __restrict__ rad_feat) {
  int e = blockIdx.x * blockDim.x + threadIdx.x;
  if (e >= Ee) return;
  int row = el[e * 3 + 0], col = el[e * 3 + 1];
  float xr[Cc * 3], xc[Cc * 3];
#pragma unroll
  for (int i = 0; i < Cc * 3; ++i) { xr[i] = x[row * Cc * 3 + i]; xc[i] = x[col * Cc * 3 + i]; }
  float wr[Cc], wc[Cc];
#pragma unroll
  for (int c = 0; c < Cc; ++c) { wr[c] = cw[row * Cc + c]; wc[c] = cw[col * Cc + c]; }
  float radial[16];
#pragma unroll
  for (int c1 = 0; c1 < Cc; ++c1) {
#pragma unroll
    for (int c2 = 0; c2 < Cc; ++c2) {
      float d0 = xr[c1 * 3 + 0] - xc[c2 * 3 + 0];
      float d1 = xr[c1 * 3 + 1] - xc[c2 * 3 + 1];
      float d2 = xr[c1 * 3 + 2] - xc[c2 * 3 + 2];
      int q = c1 * Cc + c2;
      xd[(size_t)e * 48 + q * 3 + 0] = d0;
      xd[(size_t)e * 48 + q * 3 + 1] = d1;
      xd[(size_t)e * 48 + q * 3 + 2] = d2;
      radial[q] = (d0 * d0 + d1 * d1 + d2 * d2) * wr[c1] * wc[c2];
    }
  }
#pragma unroll
  for (int f = 0; f < CFf; ++f) {
    float s = br[f];
#pragma unroll
    for (int q = 0; q < 16; ++q) s += radial[q] * Wr[q * CFf + f];
    rad_feat[(size_t)e * CFf + f] = (bf16_t)silu_f(s);
  }
}

// m_in (E x 576 bf16) = [h[row] | h[col] | rad_feat | ca[row] | ca[col] | 0]
// 8 halves per thread; segment boundaries (256/512/528/544/560) are 8-multiples.
__global__ void k_assemble_min(const int* __restrict__ el, const float* __restrict__ h,
                               const bf16_t* __restrict__ rad_feat, const float* __restrict__ ca,
                               bf16_t* __restrict__ m_in) {
  long long idx = (long long)blockIdx.x * blockDim.x + threadIdx.x;
  if (idx >= (long long)Ee * (KE1 / 8)) return;
  int e = (int)(idx / (KE1 / 8));
  int j0 = (int)(idx - (long long)e * (KE1 / 8)) * 8;
  int row = el[e * 3 + 0], col = el[e * 3 + 1];
  bf16x8 v;
  if (j0 < 256) {
    const float* s = h + (size_t)row * Hh + j0;
#pragma unroll
    for (int i = 0; i < 8; ++i) v[i] = (bf16_t)s[i];
  } else if (j0 < 512) {
    const float* s = h + (size_t)col * Hh + (j0 - 256);
#pragma unroll
    for (int i = 0; i < 8; ++i) v[i] = (bf16_t)s[i];
  } else if (j0 < 528) {
    const bf16_t* s = rad_feat + (size_t)e * CFf + (j0 - 512);
#pragma unroll
    for (int i = 0; i < 8; ++i) v[i] = s[i];
  } else if (j0 < 544) {
    const float* s = ca + row * CFf + (j0 - 528);
#pragma unroll
    for (int i = 0; i < 8; ++i) v[i] = (bf16_t)s[i];
  } else if (j0 < 560) {
    const float* s = ca + col * CFf + (j0 - 544);
#pragma unroll
    for (int i = 0; i < 8; ++i) v[i] = (bf16_t)s[i];
  } else {
#pragma unroll
    for (int i = 0; i < 8; ++i) v[i] = (bf16_t)0.0f;
  }
  *(bf16x8*)(m_in + (size_t)e * KE1 + j0) = v;
}

// ---------------- WMMA GEMM kernels ----------------
// C(M x 256) = act(A(M x K) @ W(K x 256) + bias) [* rowscale];  WT is W transposed (256 x K)
// Double-buffered LDS staging. Wave map: 2 row-groups x 4 col-slices; each wave computes a
// 32x64 patch as 2 A-frags x 4 B-frags (8 accumulators, 2 WMMAs per B fragment).
__global__ __launch_bounds__(256) void k_gemm_n256(
    const bf16_t* __restrict__ A, int lda,
    const bf16_t* __restrict__ WT,
    const float* __restrict__ bias, const float* __restrict__ rowscale,
    int M, int K, int act,
    bf16_t* __restrict__ outB, float* __restrict__ outF, int ldo) {
  __shared__ alignas(16) bf16_t As[2][64][40];
  __shared__ alignas(16) bf16_t Ws[2][256][40];
  int tid = threadIdx.x;
  int lane = tid & 31, wave = tid >> 5;
  int g = wave >> 2;        // row group: mtiles {2g, 2g+1}
  int nbase = (wave & 3) * 64;
  int row0 = blockIdx.x * 64;
  int rA = tid >> 2, cA = (tid & 3) * 8;
  const bf16_t* wrow = WT + (size_t)tid * K;

  v8f acc[8];  // acc[i*4+j]: i = local mtile (0/1), j = 16-wide ntile within the 64 slice
#pragma unroll
  for (int j = 0; j < 8; ++j)
#pragma unroll
    for (int v = 0; v < 8; ++v) acc[j][v] = 0.0f;

  auto ldA = [&](int kc) -> bf16x8 {
    bf16x8 av;
    int grow = row0 + rA;
    if (grow < M) {
      av = *(const bf16x8*)(A + (size_t)grow * lda + kc * 32 + cA);
    } else {
#pragma unroll
      for (int i = 0; i < 8; ++i) av[i] = (bf16_t)0.0f;
    }
    return av;
  };

  {  // prologue: stage chunk 0 into buffer 0
    bf16x8 av = ldA(0);
    *(bf16x8*)(&As[0][rA][cA]) = av;
    *(bf16x8*)(&Ws[0][tid][0])  = *(const bf16x8*)(wrow + 0);
    *(bf16x8*)(&Ws[0][tid][8])  = *(const bf16x8*)(wrow + 8);
    *(bf16x8*)(&Ws[0][tid][16]) = *(const bf16x8*)(wrow + 16);
    *(bf16x8*)(&Ws[0][tid][24]) = *(const bf16x8*)(wrow + 24);
  }
  int nchunks = K >> 5;
  int buf = 0;
  for (int kc = 0; kc < nchunks; ++kc, buf ^= 1) {
    __syncthreads();
    bool hasnext = (kc + 1 < nchunks);
    bf16x8 av, w0, w1, w2, w3;
    if (hasnext) {  // issue next chunk's global loads; latency hidden under WMMAs
      av = ldA(kc + 1);
      const bf16_t* s = wrow + (kc + 1) * 32;
      w0 = *(const bf16x8*)(s + 0);
      w1 = *(const bf16x8*)(s + 8);
      w2 = *(const bf16x8*)(s + 16);
      w3 = *(const bf16x8*)(s + 24);
    }
    v16bf a0 = load_a_frag(&As[buf][(g * 2 + 0) * 16][0], lane);
    v16bf a1 = load_a_frag(&As[buf][(g * 2 + 1) * 16][0], lane);
    v16bf b = load_b_frag(&Ws[buf][0][0], nbase, lane);
#pragma unroll
    for (int j = 0; j < 4; ++j) {  // load B(j+1) ahead; 2 WMMAs of shadow per load
      v16bf bn = b;
      if (j < 3) bn = load_b_frag(&Ws[buf][0][0], nbase + (j + 1) * 16, lane);
      acc[j]     = wmma_bf16(a0, b, acc[j]);
      acc[4 + j] = wmma_bf16(a1, b, acc[4 + j]);
      b = bn;
    }
    if (hasnext) {  // store staged regs into the other buffer
      int nb = buf ^ 1;
      *(bf16x8*)(&As[nb][rA][cA]) = av;
      *(bf16x8*)(&Ws[nb][tid][0])  = w0;
      *(bf16x8*)(&Ws[nb][tid][8])  = w1;
      *(bf16x8*)(&Ws[nb][tid][16]) = w2;
      *(bf16x8*)(&Ws[nb][tid][24]) = w3;
    }
  }
  int ncol = lane & 15;
#pragma unroll
  for (int i = 0; i < 2; ++i) {
    int mloc = (g * 2 + i) * 16 + ((lane >> 4) ? 8 : 0);
#pragma unroll
    for (int j = 0; j < 4; ++j) {
      int n = nbase + j * 16 + ncol;
      float bi = bias ? bias[n] : 0.0f;
#pragma unroll
      for (int v = 0; v < 8; ++v) {
        int grow = row0 + mloc + v;
        if (grow < M) {
          float val = acc[i * 4 + j][v] + bi;
          if (act) val = silu_f(val);
          if (rowscale) val *= rowscale[grow];
          if (outB) outB[(size_t)grow * ldo + n] = (bf16_t)val;
          if (outF) outF[(size_t)grow * ldo + n] = val;
        }
      }
    }
  }
}

// C(M x 16) = A(M x K) @ W(K x 16); WT is (16 x K). Block 128 rows, wave w -> mtile w.
__global__ __launch_bounds__(256) void k_gemm_n16(
    const bf16_t* __restrict__ A, int lda,
    const bf16_t* __restrict__ WT,
    int M, int K, float* __restrict__ outF) {
  __shared__ alignas(16) bf16_t As[2][128][40];
  __shared__ alignas(16) bf16_t Ws[2][16][40];
  int tid = threadIdx.x;
  int lane = tid & 31, wave = tid >> 5;
  int row0 = blockIdx.x * 128;
  int rA = tid >> 1, cA = (tid & 1) * 16;

  v8f acc;
#pragma unroll
  for (int v = 0; v < 8; ++v) acc[v] = 0.0f;

  auto ldA2 = [&](int kc, bf16x8& a0, bf16x8& a1) {
    int grow = row0 + rA;
    if (grow < M) {
      const bf16_t* src = A + (size_t)grow * lda + kc * 32 + cA;
      a0 = *(const bf16x8*)(src);
      a1 = *(const bf16x8*)(src + 8);
    } else {
#pragma unroll
      for (int i = 0; i < 8; ++i) { a0[i] = (bf16_t)0.0f; a1[i] = (bf16_t)0.0f; }
    }
  };

  {  // prologue
    bf16x8 a0, a1;
    ldA2(0, a0, a1);
    *(bf16x8*)(&As[0][rA][cA]) = a0;
    *(bf16x8*)(&As[0][rA][cA + 8]) = a1;
    if (tid < 64) {
      int n = tid >> 2, c8 = (tid & 3) * 8;
      *(bf16x8*)(&Ws[0][n][c8]) = *(const bf16x8*)(WT + (size_t)n * K + c8);
    }
  }
  int nchunks = K >> 5;
  int buf = 0;
  for (int kc = 0; kc < nchunks; ++kc, buf ^= 1) {
    __syncthreads();
    bool hasnext = (kc + 1 < nchunks);
    bf16x8 a0, a1, w;
    if (hasnext) {
      ldA2(kc + 1, a0, a1);
      if (tid < 64) {
        int n = tid >> 2, c8 = (tid & 3) * 8;
        w = *(const bf16x8*)(WT + (size_t)n * K + (kc + 1) * 32 + c8);
      }
    }
    v16bf afrag = load_a_frag(&As[buf][wave * 16][0], lane);
    v16bf bfrag = load_b_frag(&Ws[buf][0][0], 0, lane);
    acc = wmma_bf16(afrag, bfrag, acc);
    if (hasnext) {
      int nb = buf ^ 1;
      *(bf16x8*)(&As[nb][rA][cA]) = a0;
      *(bf16x8*)(&As[nb][rA][cA + 8]) = a1;
      if (tid < 64) {
        int n = tid >> 2, c8 = (tid & 3) * 8;
        *(bf16x8*)(&Ws[nb][n][c8]) = w;
      }
    }
  }
  int mloc = wave * 16 + ((lane >> 4) ? 8 : 0);
  int ncol = lane & 15;
#pragma unroll
  for (int v = 0; v < 8; ++v) {
    int grow = row0 + mloc + v;
    if (grow < M) outF[(size_t)grow * 16 + ncol] = acc[v];
  }
}

// ---------------- scatter / update kernels ----------------

__global__ void k_scatter(const int* __restrict__ el, const bf16_t* __restrict__ m,
                          const float* __restrict__ xd, const float* __restrict__ phi,
                          float* __restrict__ agg, float* __restrict__ x_acc,
                          float* __restrict__ cnt) {
  int e = blockIdx.x;
  int tid = threadIdx.x;
  int row = el[e * 3 + 0], col = el[e * 3 + 1], rel = el[e * 3 + 2];
  float mv = (float)m[(size_t)e * Hh + tid];
  atomicAdd(&agg[((size_t)col * Rr + rel) * Hh + tid], mv);
  if (tid < 12) {
    int c = tid / 3, j = tid - c * 3;
    float s = 0.0f;
#pragma unroll
    for (int c2 = 0; c2 < Cc; ++c2)
      s += xd[(size_t)e * 48 + (c * Cc + c2) * 3 + j] * phi[(size_t)e * 16 + c * Cc + c2];
    atomicAdd(&x_acc[row * 12 + tid], s * 0.25f);
  }
  if (tid == 12) atomicAdd(&cnt[row], 1.0f);
}

__global__ void k_x_update(const float* __restrict__ x_in, const float* __restrict__ x_acc,
                           const float* __restrict__ cnt, float* __restrict__ x_out) {
  int i = blockIdx.x * blockDim.x + threadIdx.x;
  if (i >= Nn * 12) return;
  float c = fmaxf(cnt[i / 12], 1.0f);
  x_out[i] = x_in[i] + x_acc[i] / c;
}

// Anode (N x 2048 bf16) = [agg (1792) | h (256)], 8 halves per thread
__global__ void k_assemble_node(const float* __restrict__ agg, const float* __restrict__ h,
                                bf16_t* __restrict__ Anode) {
  long long idx = (long long)blockIdx.x * blockDim.x + threadIdx.x;
  if (idx >= (long long)Nn * (KN1 / 8)) return;
  int n = (int)(idx / (KN1 / 8));
  int j0 = (int)(idx - (long long)n * (KN1 / 8)) * 8;
  const float* s = (j0 < Rr * Hh) ? (agg + (size_t)n * (Rr * Hh) + j0)
                                  : (h + (size_t)n * Hh + (j0 - Rr * Hh));
  bf16x8 v;
#pragma unroll
  for (int i = 0; i < 8; ++i) v[i] = (bf16_t)s[i];
  *(bf16x8*)(Anode + (size_t)n * KN1 + j0) = v;
}

// LayerNorm over H=256, one wave per node
__global__ __launch_bounds__(256) void k_layernorm(const float* __restrict__ hp,
                                                   const float* __restrict__ g,
                                                   const float* __restrict__ b,
                                                   float* __restrict__ out) {
  int lane = threadIdx.x & 31, wave = threadIdx.x >> 5;
  int node = blockIdx.x * 8 + wave;
  if (node >= Nn) return;
  float v[8], s = 0.0f, sq = 0.0f;
#pragma unroll
  for (int i = 0; i < 8; ++i) {
    v[i] = hp[(size_t)node * Hh + i * 32 + lane];
    s += v[i];
    sq += v[i] * v[i];
  }
#pragma unroll
  for (int m = 16; m >= 1; m >>= 1) {
    s += __shfl_xor(s, m, 32);
    sq += __shfl_xor(sq, m, 32);
  }
  float mu = s * (1.0f / Hh);
  float var = sq * (1.0f / Hh) - mu * mu;
  float rinv = rsqrtf(var + 1e-5f);
#pragma unroll
  for (int i = 0; i < 8; ++i) {
    int c = i * 32 + lane;
    out[(size_t)node * Hh + c] = (v[i] - mu) * rinv * g[c] + b[c];
  }
}

// ---------------- host orchestration ----------------

extern "C" void kernel_launch(void* const* d_in, const int* in_sizes, int n_in,
                              void* d_out, int out_size, void* d_ws, size_t ws_size,
                              hipStream_t stream) {
  const float* in_h   = (const float*)d_in[0];
  const float* coords = (const float*)d_in[1];
  const float* attr   = (const float*)d_in[2];
  const float* cw     = (const float*)d_in[3];
  const float* ew     = (const float*)d_in[4];
  const int*   el     = (const int*)d_in[5];
  const float* W_rad  = (const float*)d_in[6];
  const float* b_rad  = (const float*)d_in[7];
  const float* W_e1   = (const float*)d_in[8];
  const float* b_e1   = (const float*)d_in[9];
  const float* W_e2   = (const float*)d_in[10];
  const float* b_e2   = (const float*)d_in[11];
  const float* W_c1   = (const float*)d_in[12];
  const float* b_c1   = (const float*)d_in[13];
  const float* W_c2   = (const float*)d_in[14];
  const float* W_n1   = (const float*)d_in[15];
  const float* b_n1   = (const float*)d_in[16];
  const float* W_n2   = (const float*)d_in[17];
  const float* b_n2   = (const float*)d_in[18];
  const float* ln_g   = (const float*)d_in[19];
  const float* ln_b   = (const float*)d_in[20];

  char* ws = (char*)d_ws;
  size_t off = 0;
  auto alloc = [&](size_t bytes) -> char* {
    off = (off + 255) & ~(size_t)255;
    char* p = ws + off;
    off += bytes;
    return p;
  };

  bf16_t* wWe1 = (bf16_t*)alloc((size_t)Ll * KE1 * 256 * 2);   // all stored transposed (N x K)
  bf16_t* wWe2 = (bf16_t*)alloc((size_t)Ll * 256 * 256 * 2);
  bf16_t* wWc1 = (bf16_t*)alloc((size_t)Ll * 256 * 256 * 2);
  bf16_t* wWc2 = (bf16_t*)alloc((size_t)Ll * 256 * 16 * 2);
  bf16_t* wWn1 = (bf16_t*)alloc((size_t)Ll * KN1 * 256 * 2);
  bf16_t* wWn2 = (bf16_t*)alloc((size_t)Ll * 256 * 256 * 2);
  float*  ca    = (float*)alloc((size_t)Nn * CFf * 4);
  float*  xd    = (float*)alloc((size_t)Ee * 48 * 4);
  bf16_t* radT3 = (bf16_t*)alloc((size_t)Ee * CFf * 2);          // alias: rad_feat / T3
  bf16_t* minA  = (bf16_t*)alloc((size_t)Ee * KE1 * 2);          // alias: m_in / Anode
  bf16_t* T1    = (bf16_t*)alloc((size_t)Ee * Hh * 2);           // alias: T1 / T2
  bf16_t* mB    = (bf16_t*)alloc((size_t)Ee * Hh * 2);
  float*  phi   = (float*)alloc((size_t)Ee * 16 * 4);
  float*  agg   = (float*)alloc((size_t)Nn * Rr * Hh * 4);
  float*  x_acc = (float*)alloc((size_t)Nn * 12 * 4);
  float*  cnt   = (float*)alloc((size_t)Nn * 4);
  float*  hA    = (float*)alloc((size_t)Nn * Hh * 4);
  float*  xA    = (float*)alloc((size_t)Nn * 12 * 4);
  float*  hpre  = (float*)alloc((size_t)Nn * Hh * 4);

  // ---- one-time: transpose+convert weights to bf16, compute ca ----
  auto cvtT = [&](const float* s, bf16_t* d, int K, int Nc) {
    long long n = (long long)K * Nc;
    k_cvtT<<<(unsigned)((n + 255) / 256), 256, 0, stream>>>(s, d, K, Nc);
  };
  for (int l = 0; l < Ll; ++l) {
    cvtT(W_e2 + (size_t)l * 256 * 256, wWe2 + (size_t)l * 256 * 256, 256, 256);
    cvtT(W_c1 + (size_t)l * 256 * 256, wWc1 + (size_t)l * 256 * 256, 256, 256);
    cvtT(W_c2 + (size_t)l * 256 * 16,  wWc2 + (size_t)l * 256 * 16,  256, 16);
    cvtT(W_n1 + (size_t)l * KN1 * 256, wWn1 + (size_t)l * KN1 * 256, KN1, 256);
    cvtT(W_n2 + (size_t)l * 256 * 256, wWn2 + (size_t)l * 256 * 256, 256, 256);
    k_pad_we1T<<<(256 * KE1 + 255) / 256, 256, 0, stream>>>(W_e1 + (size_t)l * 560 * 256,
                                                            wWe1 + (size_t)l * KE1 * 256);
  }
  k_ca<<<(Nn + 255) / 256, 256, 0, stream>>>(attr, cw, ca);

  float* out_h = (float*)d_out;
  float* out_x = (float*)d_out + (size_t)Nn * Hh;

  for (int l = 0; l < Ll; ++l) {
    const float* hsrc = (l == 0) ? in_h : hA;
    const float* xsrc = (l == 0) ? coords : xA;
    float* hdst = (l == Ll - 1) ? out_h : hA;
    float* xdst = (l == Ll - 1) ? out_x : xA;

    k_edge_geom<<<(Ee + 255) / 256, 256, 0, stream>>>(
        el, xsrc, cw, W_rad + (size_t)l * 256, b_rad + (size_t)l * 16, xd, radT3);

    k_assemble_min<<<(unsigned)(((long long)Ee * (KE1 / 8) + 255) / 256), 256, 0, stream>>>(
        el, hsrc, radT3, ca, minA);

    // T1 = silu(m_in @ W_e1 + b_e1)
    k_gemm_n256<<<Ee / 64, 256, 0, stream>>>(minA, KE1, wWe1 + (size_t)l * KE1 * 256,
                                             b_e1 + (size_t)l * 256, nullptr,
                                             Ee, KE1, 1, T1, nullptr, Hh);
    // m = silu(T1 @ W_e2 + b_e2) * ew
    k_gemm_n256<<<Ee / 64, 256, 0, stream>>>(T1, Hh, wWe2 + (size_t)l * 256 * 256,
                                             b_e2 + (size_t)l * 256, ew,
                                             Ee, Hh, 1, mB, nullptr, Hh);
    // T2 = silu(m @ W_c1 + b_c1)   (T2 aliases T1)
    k_gemm_n256<<<Ee / 64, 256, 0, stream>>>(mB, Hh, wWc1 + (size_t)l * 256 * 256,
                                             b_c1 + (size_t)l * 256, nullptr,
                                             Ee, Hh, 1, T1, nullptr, Hh);
    // phi = T2 @ W_c2
    k_gemm_n16<<<Ee / 128, 256, 0, stream>>>(T1, Hh, wWc2 + (size_t)l * 256 * 16, Ee, Hh, phi);

    // zero accumulators (agg..cnt span; inter-buffer gaps harmless)
    {
      long long span = (long long)(((char*)cnt + Nn * 4) - (char*)agg) / 4;
      k_zero<<<4096, 256, 0, stream>>>(agg, span);
    }
    k_scatter<<<Ee, 256, 0, stream>>>(el, mB, xd, phi, agg, x_acc, cnt);
    k_x_update<<<(Nn * 12 + 255) / 256, 256, 0, stream>>>(xsrc, x_acc, cnt, xdst);

    // node path
    k_assemble_node<<<(unsigned)(((long long)Nn * (KN1 / 8) + 255) / 256), 256, 0, stream>>>(
        agg, hsrc, minA);
    // T3 = silu(Anode @ W_n1 + b_n1)   (T3 aliases rad_feat)
    k_gemm_n256<<<(Nn + 63) / 64, 256, 0, stream>>>(minA, KN1, wWn1 + (size_t)l * KN1 * 256,
                                                    b_n1 + (size_t)l * 256, nullptr,
                                                    Nn, KN1, 1, radT3, nullptr, Hh);
    // h_pre = T3 @ W_n2 + b_n2
    k_gemm_n256<<<(Nn + 63) / 64, 256, 0, stream>>>(radT3, Hh, wWn2 + (size_t)l * 256 * 256,
                                                    b_n2 + (size_t)l * 256, nullptr,
                                                    Nn, Hh, 0, nullptr, hpre, Hh);
    k_layernorm<<<(Nn + 7) / 8, 256, 0, stream>>>(hpre, ln_g + (size_t)l * 256,
                                                  ln_b + (size_t)l * 256, hdst);
  }
}
